// AttentionPairBias_21440476741693
// MI455X (gfx1250) — matrix operand order repaired and checked by reading, not compiled, not run on hardware
//
#include <hip/hip_runtime.h>
#include <hip/hip_bf16.h>

// ---------------------------------------------------------------------------
// AttentionPairBias for gfx1250 (MI455X). B=1, S=1024, D=512, H=16, HD=32,
// CZ=128.  Bandwidth floor is the 512MB read of z (~22us @ 23.3 TB/s); all
// matrix math uses v_wmma_f32_16x16x32_bf16 (fp32 accumulate).
// Round 2: bias stored bf16 (halves bias traffic, packed 16B stores),
// V stored transposed [H][HD][S] so PV B-tiles load as contiguous 32B.
// ---------------------------------------------------------------------------

#define S_LEN 1024
#define DD    512
#define NH    16
#define HDIM  32
#define CZ    128

typedef __attribute__((ext_vector_type(16))) __bf16 v16bf;
typedef __attribute__((ext_vector_type(8)))  __bf16 v8bf;
typedef __attribute__((ext_vector_type(8)))  float  v8f;

static __device__ inline v8f wmma_bf16(v16bf a, v16bf b, v8f c) {
  // D = A(16x32) * B(32x16) + C, fp32 accumulate
  return __builtin_amdgcn_wmma_f32_16x16x32_bf16(
      false, a, false, b, (short)0, c, false, false);
}

// A-tile (16 rows x 32 K) from fp32 row-major, converting to bf16.
// Contiguous K order: lane l(0..15)/l+16 hold row l; K = hi*16 + 2vg, +1 in VGPR vg.
static __device__ inline v16bf load_a_f32(const float* __restrict__ src,
                                          int stride, int lane) {
  int l = lane & 15, hi = lane >> 4;
  const float* row = src + (size_t)l * stride;
  v16bf a;
#pragma unroll
  for (int vg = 0; vg < 8; ++vg) {
    int k = 2 * vg + hi * 16;
    a[2 * vg]     = (__bf16)row[k];
    a[2 * vg + 1] = (__bf16)row[k + 1];
  }
  return a;
}

static __device__ inline v16bf load_a_bf16(const __bf16* __restrict__ src,
                                           int stride, int lane) {
  int l = lane & 15, hi = lane >> 4;
  const __bf16* row = src + (size_t)l * stride;
  v16bf a;
#pragma unroll
  for (int vg = 0; vg < 8; ++vg) {
    int k = 2 * vg + hi * 16;
    a[2 * vg]     = row[k];
    a[2 * vg + 1] = row[k + 1];
  }
  return a;
}

// B-tile (32 K x 16 N) where B[k][n] = W[n*stride + k] (i.e. op is x @ W^T).
// Contiguous K order (as load_a_*): lane n holds column n; VGPR vg holds K = hi*16 + 2vg, +1.
static __device__ inline v16bf load_bT_f32(const float* __restrict__ W,
                                           int stride, int lane) {
  int n = lane & 15, hi = lane >> 4;
  const float* col = W + (size_t)n * stride + hi * 16;
  v16bf b;
#pragma unroll
  for (int vg = 0; vg < 8; ++vg) {
    b[2 * vg]     = (__bf16)col[2 * vg];
    b[2 * vg + 1] = (__bf16)col[2 * vg + 1];
  }
  return b;
}

static __device__ inline v16bf load_bT_bf16(const __bf16* __restrict__ W,
                                            int stride, int lane) {
  int n = lane & 15, hi = lane >> 4;
  const __bf16* col = W + (size_t)n * stride + hi * 16;
  v16bf b;
#pragma unroll
  for (int vg = 0; vg < 8; ++vg) {
    b[2 * vg]     = col[2 * vg];
    b[2 * vg + 1] = col[2 * vg + 1];
  }
  return b;
}

// Channel held by this lane for A-chunk element index idx (0..63), per the
// contiguous K order used here (K = 32c + 16hi + i).
static __device__ inline int chan_of(int idx, int hi) {
  int c = idx >> 4, i = idx & 15;
  return c * 32 + hi * 16 + i;
}

// ---------------------------------------------------------------------------
// Kernel 1: q = (s@wq^T + bq) * HD^-0.5  (bf16, [S][D])
//           k = s@wk^T (bf16, [S][D])
//           v = s@wv^T (bf16, TRANSPOSED to [H][HD][S])
//           g = sigmoid(s@wg^T) (f32, [S][D])
// One wave per (matrix, 16-row tile, 16-col tile).
// ---------------------------------------------------------------------------
__global__ __launch_bounds__(128) void k_proj(
    const float* __restrict__ s, const float* __restrict__ wq,
    const float* __restrict__ bq, const float* __restrict__ wk,
    const float* __restrict__ wv, const float* __restrict__ wg,
    __bf16* __restrict__ qb, __bf16* __restrict__ kb,
    __bf16* __restrict__ vbt, float* __restrict__ gf) {
  int wid  = blockIdx.x * 4 + (threadIdx.x >> 5);
  int lane = threadIdx.x & 31;
  int mat  = wid >> 11;          // 0..3 : q,k,v,g
  int rem  = wid & 2047;
  int rt   = rem >> 5;           // row tile 0..63
  int ct   = rem & 31;           // col tile 0..31
  const float* W = (mat == 0) ? wq : (mat == 1) ? wk : (mat == 2) ? wv : wg;
  const float* arow = s + (size_t)rt * 16 * DD;
  const float* wrow = W + (size_t)ct * 16 * DD;

  v8f acc = {};
#pragma unroll 4
  for (int kk = 0; kk < DD; kk += 32) {
    v16bf a = load_a_f32(arow + kk, DD, lane);
    v16bf b = load_bT_f32(wrow + kk, DD, lane);
    acc = wmma_bf16(a, b, acc);
  }

  int l = lane & 15, hi = lane >> 4;
  int col = ct * 16 + l;
  if (mat == 0) {
    float bv = bq[col];
    const float sc = 0.17677669529663687f;  // 1/sqrt(32)
#pragma unroll
    for (int r = 0; r < 8; ++r) {
      int row = rt * 16 + r + 8 * hi;
      qb[(size_t)row * DD + col] = (__bf16)((acc[r] + bv) * sc);
    }
  } else if (mat == 1) {
#pragma unroll
    for (int r = 0; r < 8; ++r) {
      int row = rt * 16 + r + 8 * hi;
      kb[(size_t)row * DD + col] = (__bf16)acc[r];
    }
  } else if (mat == 2) {
    // transpose: vbt[h][d][kpos], kpos = row.  8 consecutive rows per lane
    // -> one packed 16B store.
    int h = col >> 5, d = col & 31;
    v8bf pk;
#pragma unroll
    for (int r = 0; r < 8; ++r) pk[r] = (__bf16)acc[r];
    __bf16* base = vbt + ((size_t)h * HDIM + d) * S_LEN + rt * 16 + 8 * hi;
    *(v8bf*)base = pk;
  } else {
#pragma unroll
    for (int r = 0; r < 8; ++r) {
      int row = rt * 16 + r + 8 * hi;
      gf[(size_t)row * DD + col] = 1.0f / (1.0f + __expf(-acc[r]));
    }
  }
}

// ---------------------------------------------------------------------------
// Kernel 2 (bandwidth-dominant): per pair (q,k): layernorm over CZ=128 then
// 128->16 head projection, via WMMA.  One wave handles 16 k-positions of one
// q: A = zn tile [16 pairs x 128ch], B = wz^T [128 x 16 heads].
// bias layout: [H][S][S] bf16, packed 16B store per lane.
// ---------------------------------------------------------------------------
__global__ __launch_bounds__(128) void k_bias(
    const float* __restrict__ z, const float* __restrict__ znw,
    const float* __restrict__ znb, const float* __restrict__ wz,
    __bf16* __restrict__ bias) {
  int wid  = blockIdx.x * 4 + (threadIdx.x >> 5);
  int lane = threadIdx.x & 31;
  int l = lane & 15, hi = lane >> 4;
  int q  = wid >> 6;
  int k0 = (wid & 63) << 4;

  const float* zrow = z + ((size_t)q * S_LEN + (size_t)(k0 + l)) * CZ;
  __builtin_prefetch(zrow + 16 * CZ, 0, 0);  // next k-tile, same lane slot

  // Gather this lane's 64 channels (A-layout order) + stats
  float zv[64];
  float sum = 0.f, sq = 0.f;
#pragma unroll
  for (int i = 0; i < 64; ++i) {
    float v = zrow[chan_of(i, hi)];
    zv[i] = v;
    sum += v;
    sq  += v * v;
  }
  sum += __shfl_xor(sum, 16, 32);  // partner lane holds the other 64 channels
  sq  += __shfl_xor(sq, 16, 32);
  float mean = sum * (1.0f / 128.0f);
  float var  = sq * (1.0f / 128.0f) - mean * mean;
  float rstd = rsqrtf(var + 1e-5f);

  v8f acc = {};
#pragma unroll
  for (int c = 0; c < 4; ++c) {
    v16bf a, b;
#pragma unroll
    for (int i = 0; i < 16; ++i) {
      int cc = chan_of(c * 16 + i, hi);
      a[i] = (__bf16)(((zv[c * 16 + i] - mean) * rstd) * znw[cc] + znb[cc]);
      // B[k=ch][n=head] = wz[head*CZ + ch]; lane n=l, K = c*32 + hi*16 + i
      b[i] = (__bf16)wz[l * CZ + c * 32 + hi * 16 + i];
    }
    acc = wmma_bf16(a, b, acc);
  }

  // tile rows M = k-offset (8 consecutive per lane-half), cols N = head(=l):
  // one packed 16B store.
  v8bf pk;
#pragma unroll
  for (int r = 0; r < 8; ++r) pk[r] = (__bf16)acc[r];
  *(v8bf*)(bias + (size_t)l * S_LEN * S_LEN + (size_t)q * S_LEN + k0 + 8 * hi) = pk;
}

// ---------------------------------------------------------------------------
// Kernel 3: flash attention.  One wave per (head, 16-row q tile); streams all
// 1024 k in chunks of 32 with online softmax.  QK^T and PV both WMMA; P is
// re-laid out C->A via a 1KB LDS round trip (same-wave, in-order DS).
// V is read from the [H][HD][S] transposed layout: contiguous 32B per lane.
// ---------------------------------------------------------------------------
__global__ __launch_bounds__(128) void k_attn(
    const __bf16* __restrict__ qb, const __bf16* __restrict__ kb,
    const __bf16* __restrict__ vbt, const __bf16* __restrict__ bias,
    float* __restrict__ of) {
  __shared__ __bf16 pt[4][16 * 32];
  int wslot = threadIdx.x >> 5;
  int wid   = blockIdx.x * 4 + wslot;
  int lane  = threadIdx.x & 31;
  int l = lane & 15, hi = lane >> 4;
  int h  = wid >> 6;
  int q0 = (wid & 63) << 4;

  v16bf aq = load_a_bf16(qb + (size_t)q0 * DD + h * HDIM, DD, lane);

  float mrun[8], lrun[8];
#pragma unroll
  for (int r = 0; r < 8; ++r) { mrun[r] = -3.0e38f; lrun[r] = 0.f; }
  v8f acc0 = {}, acc1 = {};
  v8f zf = {};

  const __bf16* brow = bias + (size_t)h * S_LEN * S_LEN + (size_t)q0 * S_LEN;
  // V B-tile bases: head-dim d = l (acc0 half) and l+16 (acc1 half)
  const __bf16* vrow0 = vbt + ((size_t)h * HDIM + l) * S_LEN + hi * 16;
  const __bf16* vrow1 = vrow0 + (size_t)16 * S_LEN;
  __bf16* myp = pt[wslot];

  for (int k0 = 0; k0 < S_LEN; k0 += 32) {
    // logits for 32 k-positions (two 16x16 tiles), scale pre-folded into q
    v16bf bk0 = load_bT_bf16(kb + (size_t)k0 * DD + h * HDIM, DD, lane);
    v16bf bk1 = load_bT_bf16(kb + (size_t)(k0 + 16) * DD + h * HDIM, DD, lane);
    v8f s0 = wmma_bf16(aq, bk0, zf);
    v8f s1 = wmma_bf16(aq, bk1, zf);

#pragma unroll
    for (int r = 0; r < 8; ++r) {
      const __bf16* br = brow + (size_t)(r + 8 * hi) * S_LEN + k0 + l;
      s0[r] += (float)br[0];
      s1[r] += (float)br[16];
    }

    // online softmax per row (row lives in one 16-lane half)
#pragma unroll
    for (int r = 0; r < 8; ++r) {
      float t = fmaxf(s0[r], s1[r]);
#pragma unroll
      for (int off = 8; off >= 1; off >>= 1) t = fmaxf(t, __shfl_xor(t, off, 16));
      float mnew  = fmaxf(mrun[r], t);
      float scale = __expf(mrun[r] - mnew);
      float p0 = __expf(s0[r] - mnew);
      float p1 = __expf(s1[r] - mnew);
      float ps = p0 + p1;
#pragma unroll
      for (int off = 8; off >= 1; off >>= 1) ps += __shfl_xor(ps, off, 16);
      lrun[r] = lrun[r] * scale + ps;
      mrun[r] = mnew;
      acc0[r] *= scale;
      acc1[r] *= scale;
      myp[(r + 8 * hi) * 32 + l]      = (__bf16)p0;
      myp[(r + 8 * hi) * 32 + 16 + l] = (__bf16)p1;
    }

    // reload P in A layout (16 q rows x 32 k)
    v16bf ap;
#pragma unroll
    for (int vg = 0; vg < 8; ++vg) {
      int k = 2 * vg + hi * 16;
      ap[2 * vg]     = myp[l * 32 + k];
      ap[2 * vg + 1] = myp[l * 32 + k + 1];
    }

    // V tiles from transposed layout: b[i] = v[k0 + hi*16 + i][d]
    // = vbt[h][d][k0 + hi*16 + i] -> 16 consecutive bf16 (32B aligned).
    v16bf bv0 = *(const v16bf*)(vrow0 + k0);
    v16bf bv1 = *(const v16bf*)(vrow1 + k0);
    acc0 = wmma_bf16(ap, bv0, acc0);
    acc1 = wmma_bf16(ap, bv1, acc1);
  }

#pragma unroll
  for (int r = 0; r < 8; ++r) {
    float inv = 1.0f / lrun[r];
    float* orow = of + (size_t)(q0 + r + 8 * hi) * DD + h * HDIM + l;
    orow[0]  = acc0[r] * inv;
    orow[16] = acc1[r] * inv;
  }
}

// ---------------------------------------------------------------------------
// Kernel 4: out = (o * g) @ wo^T   (fp32 out)
// ---------------------------------------------------------------------------
__global__ __launch_bounds__(128) void k_out(
    const float* __restrict__ of, const float* __restrict__ gf,
    const float* __restrict__ wo, float* __restrict__ out) {
  int wid  = blockIdx.x * 4 + (threadIdx.x >> 5);
  int lane = threadIdx.x & 31;
  int rt = wid >> 5, ct = wid & 31;
  int l = lane & 15, hi = lane >> 4;

  const float* orl = of + ((size_t)rt * 16 + l) * DD;
  const float* grl = gf + ((size_t)rt * 16 + l) * DD;
  const float* wrow = wo + (size_t)ct * 16 * DD;

  v8f acc = {};
#pragma unroll 4
  for (int kk = 0; kk < DD; kk += 32) {
    v16bf a;
#pragma unroll
    for (int vg = 0; vg < 8; ++vg) {
      int k = 2 * vg + hi * 16 + kk;
      a[2 * vg]     = (__bf16)(orl[k] * grl[k]);
      a[2 * vg + 1] = (__bf16)(orl[k + 1] * grl[k + 1]);
    }
    v16bf b = load_bT_f32(wrow + kk, DD, lane);
    acc = wmma_bf16(a, b, acc);
  }

  int col = ct * 16 + l;
#pragma unroll
  for (int r = 0; r < 8; ++r)
    out[(size_t)(rt * 16 + r + 8 * hi) * DD + col] = acc[r];
}

// ---------------------------------------------------------------------------
extern "C" void kernel_launch(void* const* d_in, const int* in_sizes, int n_in,
                              void* d_out, int out_size, void* d_ws,
                              size_t ws_size, hipStream_t stream) {
  const float* s   = (const float*)d_in[0];
  const float* z   = (const float*)d_in[1];
  const float* wq  = (const float*)d_in[2];
  const float* bq  = (const float*)d_in[3];
  const float* wk  = (const float*)d_in[4];
  const float* wv  = (const float*)d_in[5];
  const float* wg  = (const float*)d_in[6];
  const float* znw = (const float*)d_in[7];
  const float* znb = (const float*)d_in[8];
  const float* wz  = (const float*)d_in[9];
  const float* wo  = (const float*)d_in[10];
  float* out = (float*)d_out;

  // Workspace layout (byte offsets; 39 MB total)
  char* w = (char*)d_ws;
  __bf16* qb   = (__bf16*)(w + 0);                      // 1 MB  [S][D]
  __bf16* kb   = (__bf16*)(w + (1u << 20));             // 1 MB  [S][D]
  __bf16* vbt  = (__bf16*)(w + (2u << 20));             // 1 MB  [H][HD][S]
  float*  gf   = (float*)(w + (3u << 20));              // 2 MB  [S][D]
  float*  of   = (float*)(w + (5u << 20));              // 2 MB  [S][D]
  __bf16* bias = (__bf16*)(w + (7u << 20));             // 32 MB [H][S][S]

  // 4 matrices * 64 row tiles * 32 col tiles = 8192 waves, 4 waves/block
  k_proj<<<2048, 128, 0, stream>>>(s, wq, bq, wk, wv, wg, qb, kb, vbt, gf);
  // 1024 q * 64 k-tiles = 65536 waves (reads the 512MB z stream)
  k_bias<<<16384, 128, 0, stream>>>(z, znw, znb, wz, bias);
  // 16 heads * 64 q-tiles = 1024 waves
  k_attn<<<256, 128, 0, stream>>>(qb, kb, vbt, bias, of);
  // 64 row tiles * 32 col tiles = 2048 waves
  k_out<<<512, 128, 0, stream>>>(of, gf, wo, out);
}
